// SparseMoE_8538394984494
// MI455X (gfx1250) — matrix operand order repaired and checked by reading, not compiled, not run on hardware
//
#include <hip/hip_runtime.h>
#include <hip/hip_bf16.h>

typedef __attribute__((ext_vector_type(16))) _Float16 v16h;
typedef __attribute__((ext_vector_type(8)))  float    v8f;

#define T_TOK   32768   // 4 * 8192
#define D_MODEL 256
#define N_EXP   16
#define HIDDEN  512

union FragU { uint4 q[2]; v16h v; };

// ---------------------------------------------------------------------------
// x (fp32, coalesced) -> f16
// ---------------------------------------------------------------------------
__global__ void cvt_x_kernel(const float* __restrict__ x, _Float16* __restrict__ xh, int total) {
    int gid = blockIdx.x * 256 + threadIdx.x;
    if (gid < total) xh[gid] = (_Float16)x[gid];
}

// ---------------------------------------------------------------------------
// Pack a weight tensor [E][K][N] (row-major fp32) into WMMA B-fragment layout:
// dst[(((e*NT + nt)*KS + ks)*32 + lane)*16 + i] = w[e][ks*32 + (lane>=16?16:0) + i][nt*16 + (lane&15)]
// contiguous k order used here (A uses the same): 16-bit B 32x16 — lanes 0-15 hold K=0..15 (2 per VGPR),
// lanes 16-31 hold K=16..31; column = lane&15.
// ---------------------------------------------------------------------------
__global__ void pack_b_kernel(const float* __restrict__ src, _Float16* __restrict__ dst,
                              int K, int N, int total) {
    int gid = blockIdx.x * 256 + threadIdx.x;
    if (gid >= total) return;
    int i    = gid & 15;
    int lane = (gid >> 4) & 31;
    int t2   = gid >> 9;
    int KS   = K >> 5;
    int ks   = t2 % KS;
    int t3   = t2 / KS;
    int NT   = N >> 4;
    int nt   = t3 % NT;
    int e    = t3 / NT;
    int k = ks * 32 + ((lane >> 4) << 4) + i;
    int n = nt * 16 + (lane & 15);
    dst[gid] = (_Float16)src[((size_t)e * K + k) * N + n];
}

// ---------------------------------------------------------------------------
// Gating: fp32 logits, top-2 (first-occurrence tie-break like lax.top_k),
// renormalized pair weight = sigmoid(l0 - l1). Append token to each winning
// expert's list via atomic counter. Output value per token is independent of
// slot order, so atomics keep d_out deterministic.
// ---------------------------------------------------------------------------
__global__ __launch_bounds__(256)
void gate_route_kernel(const float* __restrict__ x, const float* __restrict__ gw,
                       const float* __restrict__ gb, int* __restrict__ cnt,
                       int* __restrict__ idxb, float* __restrict__ wgtb) {
    __shared__ float sX[8][D_MODEL];
    __shared__ float sL[8][N_EXP];
    const int wave = threadIdx.x >> 5;
    const int lane = threadIdx.x & 31;
    const int t = blockIdx.x * 8 + wave;

    #pragma unroll
    for (int kk = 0; kk < D_MODEL / 32; ++kk)
        sX[wave][lane + kk * 32] = x[(size_t)t * D_MODEL + lane + kk * 32];
    __syncthreads();

    if (lane < N_EXP) {
        float acc = gb[lane];
        for (int d = 0; d < D_MODEL; ++d)
            acc += sX[wave][d] * gw[d * N_EXP + lane];
        sL[wave][lane] = acc;
    }
    __syncthreads();

    if (threadIdx.x < 8) {
        const int w  = threadIdx.x;
        const int tt = blockIdx.x * 8 + w;
        float l[N_EXP];
        #pragma unroll
        for (int e = 0; e < N_EXP; ++e) l[e] = sL[w][e];
        int e0 = 0; float l0 = l[0];
        #pragma unroll
        for (int e = 1; e < N_EXP; ++e) if (l[e] > l0) { l0 = l[e]; e0 = e; }
        int e1 = -1; float l1 = -3.4e38f;
        #pragma unroll
        for (int e = 0; e < N_EXP; ++e) if (e != e0 && l[e] > l1) { l1 = l[e]; e1 = e; }
        float p0 = 1.0f / (1.0f + expf(l1 - l0));   // stable: l0 >= l1
        int s0 = atomicAdd(&cnt[e0], 1);
        idxb[e0 * T_TOK + s0] = tt; wgtb[e0 * T_TOK + s0] = p0;
        int s1 = atomicAdd(&cnt[e1], 1);
        idxb[e1 * T_TOK + s1] = tt; wgtb[e1 * T_TOK + s1] = 1.0f - p0;
    }
}

// ---------------------------------------------------------------------------
// Expert SwiGLU FFN on a 32-token tile (two 16-row WMMA sub-tiles) per block.
// 4 waves / 128 threads. Each B fragment is fetched once from L2 and consumed
// by TWO WMMAs (one per sub-tile) -> halves B traffic per FLOP vs M=16.
// Stage1: h = silu(A*W1) * (A*W2); Stage2: o = h*W3, scaled + atomic scatter.
// A-fragment layout (matches packed B): lane<16 -> row=lane,
// K [0..15]; lane>=16 -> row=lane-16, K [16..31].
// C/D f32: lane<16 -> col=lane, rows 0..7; lane>=16 -> col=lane-16, rows 8..15.
// ---------------------------------------------------------------------------
__global__ __launch_bounds__(128)
void expert_ffn_kernel(const _Float16* __restrict__ xh,
                       const _Float16* __restrict__ w1f,
                       const _Float16* __restrict__ w2f,
                       const _Float16* __restrict__ w3f,
                       const int* __restrict__ cnt,
                       const int* __restrict__ idxb,
                       const float* __restrict__ wgtb,
                       float* __restrict__ out) {
    const int e    = blockIdx.y;
    const int tile = blockIdx.x;
    const int ne   = cnt[e];
    if (tile * 32 >= ne) return;   // uniform exit: EXEC all-1s at every WMMA

    __shared__ __align__(16) _Float16 sA[32 * D_MODEL];  // 16 KB
    __shared__ __align__(16) _Float16 sH[32 * HIDDEN];   // 32 KB
    __shared__ int   tokS[32];
    __shared__ float wtS[32];

    const int tid  = threadIdx.x;
    const int wave = tid >> 5;
    const int lane = tid & 31;

    if (tid < 32) {
        int p = tile * 32 + tid;
        if (p < ne) { tokS[tid] = idxb[e * T_TOK + p]; wtS[tid] = wgtb[e * T_TOK + p]; }
        else        { tokS[tid] = -1;                  wtS[tid] = 0.0f; }
    }
    __syncthreads();

    // stage x tile (32 rows x 256 halfs), vectorized b128 per lane
    {
        const uint4 z4 = make_uint4(0u, 0u, 0u, 0u);
        #pragma unroll
        for (int it = 0; it < 8; ++it) {
            int lin = it * 128 + tid;
            int row = lin >> 5;          // 32 consecutive lanes share a row
            int qc  = lin & 31;          // uint4 chunk within the row (32 * 16B = 512B)
            int tok = tokS[row];
            uint4 v = (tok >= 0) ? ((const uint4*)(xh + (size_t)tok * D_MODEL))[qc] : z4;
            ((uint4*)(sA + row * D_MODEL))[qc] = v;
        }
    }
    __syncthreads();

    const int r    = lane & 15;
    const int coff = (lane >> 4) * 16;      // A-fragment chunk base
    const int rowB = (lane >> 4) * 8;       // C-fragment row base
    const int colL = lane & 15;             // C-fragment column within tile

    // ---- Stage 1: each wave does 8 N-tiles of 16; 4 WMMAs per B pair
    for (int j = 0; j < 8; ++j) {
        const int nt = wave * 8 + j;
        v8f acc1a = {}, acc2a = {}, acc1b = {}, acc2b = {};
        #pragma unroll
        for (int ks = 0; ks < D_MODEL / 32; ++ks) {
            FragU a0, a1, b1, b2;
            const int ab = ks * 32 + coff;
            a0.q[0] = *(const uint4*)(sA + r * D_MODEL + ab);
            a0.q[1] = *(const uint4*)(sA + r * D_MODEL + ab + 8);
            a1.q[0] = *(const uint4*)(sA + (16 + r) * D_MODEL + ab);
            a1.q[1] = *(const uint4*)(sA + (16 + r) * D_MODEL + ab + 8);
            const size_t bo = (size_t)e * (D_MODEL * HIDDEN) +
                              (size_t)(((nt * (D_MODEL / 32) + ks) * 32 + lane) * 16);
            const uint4* pb1 = (const uint4*)(w1f + bo);
            b1.q[0] = pb1[0]; b1.q[1] = pb1[1];
            const uint4* pb2 = (const uint4*)(w2f + bo);
            b2.q[0] = pb2[0]; b2.q[1] = pb2[1];
            acc1a = __builtin_amdgcn_wmma_f32_16x16x32_f16(false, a0.v, false, b1.v, (short)0, acc1a, false, false);
            acc2a = __builtin_amdgcn_wmma_f32_16x16x32_f16(false, a0.v, false, b2.v, (short)0, acc2a, false, false);
            acc1b = __builtin_amdgcn_wmma_f32_16x16x32_f16(false, a1.v, false, b1.v, (short)0, acc1b, false, false);
            acc2b = __builtin_amdgcn_wmma_f32_16x16x32_f16(false, a1.v, false, b2.v, (short)0, acc2b, false, false);
        }
        const int col = nt * 16 + colL;
        #pragma unroll
        for (int m = 0; m < 8; ++m) {
            float za = acc1a[m];
            float ha = (za / (1.0f + __expf(-za))) * acc2a[m];
            sH[(rowB + m) * HIDDEN + col] = (_Float16)ha;
            float zb = acc1b[m];
            float hb = (zb / (1.0f + __expf(-zb))) * acc2b[m];
            sH[(16 + rowB + m) * HIDDEN + col] = (_Float16)hb;
        }
    }
    __syncthreads();

    // ---- Stage 2: each wave does 4 N-tiles of 16; 2 WMMAs per B fragment
    for (int j = 0; j < 4; ++j) {
        const int nt = wave * 4 + j;
        v8f acca = {}, accb = {};
        #pragma unroll
        for (int ks = 0; ks < HIDDEN / 32; ++ks) {
            FragU a0, a1, b;
            const int ab = ks * 32 + coff;
            a0.q[0] = *(const uint4*)(sH + r * HIDDEN + ab);
            a0.q[1] = *(const uint4*)(sH + r * HIDDEN + ab + 8);
            a1.q[0] = *(const uint4*)(sH + (16 + r) * HIDDEN + ab);
            a1.q[1] = *(const uint4*)(sH + (16 + r) * HIDDEN + ab + 8);
            const size_t bo = (size_t)e * (HIDDEN * D_MODEL) +
                              (size_t)(((nt * (HIDDEN / 32) + ks) * 32 + lane) * 16);
            const uint4* pb = (const uint4*)(w3f + bo);
            b.q[0] = pb[0]; b.q[1] = pb[1];
            acca = __builtin_amdgcn_wmma_f32_16x16x32_f16(false, a0.v, false, b.v, (short)0, acca, false, false);
            accb = __builtin_amdgcn_wmma_f32_16x16x32_f16(false, a1.v, false, b.v, (short)0, accb, false, false);
        }
        const int col = nt * 16 + colL;
        #pragma unroll
        for (int m = 0; m < 8; ++m) {
            const int ra = rowB + m;
            const int ta = tokS[ra];
            if (ta >= 0) atomicAdd(out + (size_t)ta * D_MODEL + col, acca[m] * wtS[ra]);
            const int rb = 16 + rowB + m;
            const int tb = tokS[rb];
            if (tb >= 0) atomicAdd(out + (size_t)tb * D_MODEL + col, accb[m] * wtS[rb]);
        }
    }
}

// ---------------------------------------------------------------------------
extern "C" void kernel_launch(void* const* d_in, const int* in_sizes, int n_in,
                              void* d_out, int out_size, void* d_ws, size_t ws_size,
                              hipStream_t stream) {
    const float* x      = (const float*)d_in[0];
    const float* gate_w = (const float*)d_in[1];
    const float* gate_b = (const float*)d_in[2];
    const float* w1     = (const float*)d_in[3];
    const float* w2     = (const float*)d_in[4];
    const float* w3     = (const float*)d_in[5];
    float* out = (float*)d_out;

    char* ws = (char*)d_ws;
    const size_t xhBytes = (size_t)T_TOK * D_MODEL * sizeof(_Float16);          // 16 MB
    const size_t wBytes  = (size_t)N_EXP * D_MODEL * HIDDEN * sizeof(_Float16); // 4 MB each
    _Float16* xh  = (_Float16*)ws;
    _Float16* w1f = (_Float16*)(ws + xhBytes);
    _Float16* w2f = (_Float16*)(ws + xhBytes + wBytes);
    _Float16* w3f = (_Float16*)(ws + xhBytes + 2 * wBytes);
    int*   cnt  = (int*)  (ws + xhBytes + 3 * wBytes);
    int*   idxb = (int*)  ((char*)cnt + 256);
    float* wgtb = (float*)((char*)idxb + (size_t)N_EXP * T_TOK * sizeof(int));

    hipMemsetAsync(out, 0, (size_t)out_size * sizeof(float), stream);
    hipMemsetAsync(cnt, 0, 256, stream);

    cvt_x_kernel<<<(T_TOK * D_MODEL) / 256, 256, 0, stream>>>(x, xh, T_TOK * D_MODEL);
    const int wTotal = N_EXP * D_MODEL * HIDDEN;
    pack_b_kernel<<<wTotal / 256, 256, 0, stream>>>(w1, w1f, D_MODEL, HIDDEN, wTotal);
    pack_b_kernel<<<wTotal / 256, 256, 0, stream>>>(w2, w2f, D_MODEL, HIDDEN, wTotal);
    pack_b_kernel<<<wTotal / 256, 256, 0, stream>>>(w3, w3f, HIDDEN, D_MODEL, wTotal);

    gate_route_kernel<<<T_TOK / 8, 256, 0, stream>>>(x, gate_w, gate_b, cnt, idxb, wgtb);

    dim3 grid(T_TOK / 32, N_EXP);   // worst-case tiles; empty tiles exit before any WMMA
    expert_ffn_kernel<<<grid, 128, 0, stream>>>(xh, w1f, w2f, w3f, cnt, idxb, wgtb, out);
}
